// CrossAttention_21784074125536
// MI455X (gfx1250) — matrix operand order repaired and checked by reading, not compiled, not run on hardware
//
#include <hip/hip_runtime.h>
#include <hip/hip_bf16.h>
#include <cstdint>
#include <cstddef>

// ---------------------------------------------------------------------------
// CDNA5 (gfx1250) cross-attention:
//   - all GEMMs + attention on v_wmma_f32_16x16x32_bf16 (fp32 accumulate)
//   - GEMM double-buffered via GLOBAL_LOAD_ASYNC_TO_LDS_B128 (ASYNCcnt)
// ---------------------------------------------------------------------------

typedef __bf16 v16bf __attribute__((ext_vector_type(16)));
typedef float  v8f   __attribute__((ext_vector_type(8)));
typedef int    v4i   __attribute__((ext_vector_type(4)));

union FragB {
    v16bf        v;
    unsigned int u[8];
};

union U4 {
    uint4          q;
    unsigned short s[8];
};

__device__ __forceinline__ unsigned short f2bf(float f) {
    unsigned int u = __float_as_uint(f);
    u += 0x7FFFu + ((u >> 16) & 1u);          // round-to-nearest-even
    return (unsigned short)(u >> 16);
}
__device__ __forceinline__ void storeC(float* p, float v)          { *p = v; }
__device__ __forceinline__ void storeC(unsigned short* p, float v) { *p = f2bf(v); }

// A-matrix (16x32 bf16) halfword-pair K index: interleaved per lane-half (ISA 7.12.2)
__device__ __forceinline__ int kpA(int vg, int half) {
    return ((vg & 4) ? (16 + 2 * (vg & 3)) : (2 * vg)) + 8 * half;
}
// B-matrix (32x16 bf16) halfword-pair K index: interleaved per lane-half (same as A)
__device__ __forceinline__ int kpB(int vg, int half) {
    return ((vg & 4) ? (16 + 2 * (vg & 3)) : (2 * vg)) + 8 * half;
}

__device__ __forceinline__ v8f wmma_bf16(const FragB& a, const FragB& b, v8f c) {
    return __builtin_amdgcn_wmma_f32_16x16x32_bf16(false, a.v, false, b.v,
                                                   (short)0, c, false, false);
}

// ---- async global->LDS copy (16B per lane), with safe fallback --------------
#if defined(__gfx1250__) && __has_builtin(__builtin_amdgcn_global_load_async_to_lds_b128)
#define USE_ASYNC_LDS 1
#else
#define USE_ASYNC_LDS 0
#endif

#if USE_ASYNC_LDS
typedef __attribute__((address_space(1))) v4i* as1_v4i_p;
typedef __attribute__((address_space(3))) v4i* as3_v4i_p;
#endif

__device__ __forceinline__ void cp_lds_b128(unsigned short* lds, const unsigned short* g) {
#if USE_ASYNC_LDS
    __builtin_amdgcn_global_load_async_to_lds_b128(
        (as1_v4i_p)(void*)g, (as3_v4i_p)(void*)lds, 0, 0);
#else
    *(uint4*)lds = *(const uint4*)g;
#endif
}

__device__ __forceinline__ void async_fence() {
#if USE_ASYNC_LDS
#if __has_builtin(__builtin_amdgcn_s_wait_asynccnt)
    __builtin_amdgcn_s_wait_asynccnt(0);
#else
    asm volatile("s_wait_asynccnt 0" ::: "memory");
#endif
#endif
}

// ---------------------------------------------------------------------------
// elementwise fp32 -> bf16 (n % 4 == 0)
// ---------------------------------------------------------------------------
__global__ __launch_bounds__(256) void convert_f32_bf16_kernel(
    const float* __restrict__ in, unsigned short* __restrict__ out, int n)
{
    int i = (blockIdx.x * 256 + threadIdx.x) * 4;
    if (i < n) {
        float4 v = *(const float4*)(in + i);
        unsigned int p0 = (unsigned int)f2bf(v.x) | ((unsigned int)f2bf(v.y) << 16);
        unsigned int p1 = (unsigned int)f2bf(v.z) | ((unsigned int)f2bf(v.w) << 16);
        uint2 p; p.x = p0; p.y = p1;
        *(uint2*)(out + i) = p;
    }
}

// ---------------------------------------------------------------------------
// W[K,N] fp32  ->  WT[N,K] bf16   (K%32==0, N%32==0)
// ---------------------------------------------------------------------------
__global__ __launch_bounds__(1024) void transpose_convert_kernel(
    const float* __restrict__ W, unsigned short* __restrict__ WT, int K, int N)
{
    __shared__ unsigned short t[32][33];
    int n0 = blockIdx.x * 32, k0 = blockIdx.y * 32;
    int tx = threadIdx.x, ty = threadIdx.y;
    t[ty][tx] = f2bf(W[(size_t)(k0 + ty) * N + n0 + tx]);
    __syncthreads();
    WT[(size_t)(n0 + ty) * K + k0 + tx] = t[tx][ty];
}

// ---------------------------------------------------------------------------
// GEMM + bias:  C[M,N] = A[M,K](bf16) * BT[N,K](bf16)^T + bias[N]
// Block 256 thr (8 waves), tile 128x128, BK=32, double-buffered async staging.
// Each wave: 32x64 -> 8 wmma per stage. M%128==0, N%128==0, K%32==0.
// ---------------------------------------------------------------------------
template <typename OT>
__global__ __launch_bounds__(256) void gemm_bf16_kernel(
    const unsigned short* __restrict__ A, const unsigned short* __restrict__ BT,
    const float* __restrict__ bias, OT* __restrict__ C,
    int M, int K, int N)
{
    constexpr int BM = 128, BN = 128, BK = 32, PAD = 40;  // 80B rows: 16B-aligned + conflict-free
    __shared__ unsigned short As[2][BM * PAD];            // [buf][m][k]
    __shared__ unsigned short Bs[2][BN * PAD];            // [buf][n][k]

    const int tid  = threadIdx.x;
    const int lane = tid & 31;
    const int wave = tid >> 5;         // 0..7
    const int wm   = wave & 3;         // 4 x 32 rows
    const int wn   = wave >> 2;        // 2 x 64 cols
    const int half = (lane >> 4) & 1;
    const int l16  = lane & 15;

    const int mBase = blockIdx.y * BM;
    const int nBase = blockIdx.x * BN;

    v8f acc[2][4] = {};                // [a-frag 16rows][b-frag 16cols]

    // stage 512 16B-chunks per tile (2 per thread per matrix)
    auto stage = [&](int buf, int k0) {
        #pragma unroll
        for (int i = 0; i < 2; ++i) {
            int c   = tid + i * 256;          // 0..511
            int row = c >> 2;
            int cc  = (c & 3) * 8;
            cp_lds_b128(&As[buf][row * PAD + cc], &A [(size_t)(mBase + row) * K + k0 + cc]);
            cp_lds_b128(&Bs[buf][row * PAD + cc], &BT[(size_t)(nBase + row) * K + k0 + cc]);
        }
    };

    stage(0, 0);
    const int T = K / BK;
    for (int kt = 0; kt < T; ++kt) {
        const int cur = kt & 1;
        async_fence();                 // this wave's pending stage(cur) done
        __syncthreads();               // all waves' stage(cur) visible; prior reads done
        if (kt + 1 < T) stage(cur ^ 1, (kt + 1) * BK);   // prefetch next while computing

        FragB a[2];
        #pragma unroll
        for (int ai = 0; ai < 2; ++ai)
            #pragma unroll
            for (int vg = 0; vg < 8; ++vg)
                a[ai].u[vg] = *(const unsigned int*)
                    &As[cur][(wm * 32 + ai * 16 + l16) * PAD + kpA(vg, half)];
        #pragma unroll
        for (int bi = 0; bi < 4; ++bi) {
            FragB b;
            #pragma unroll
            for (int vg = 0; vg < 8; ++vg)
                b.u[vg] = *(const unsigned int*)
                    &Bs[cur][(wn * 64 + bi * 16 + l16) * PAD + kpB(vg, half)];
            acc[0][bi] = wmma_bf16(a[0], b, acc[0][bi]);
            acc[1][bi] = wmma_bf16(a[1], b, acc[1][bi]);
        }
    }

    // ---- epilogue: bias + store ----
    #pragma unroll
    for (int bi = 0; bi < 4; ++bi) {
        int col  = nBase + wn * 64 + bi * 16 + l16;
        float bv = bias[col];
        #pragma unroll
        for (int ai = 0; ai < 2; ++ai)
            #pragma unroll
            for (int r = 0; r < 8; ++r) {
                size_t row = (size_t)(mBase + wm * 32 + ai * 16 + r + 8 * half);
                storeC(&C[row * (size_t)N + col], acc[ai][bi][r] + bv);
            }
    }
}

// ---------------------------------------------------------------------------
// Flash-style cross attention. Block = 128 thr (4 waves): (b, h, 64 q-rows).
// Streams G=512 keys in tiles of 64 with online softmax; bf16 WMMA, fp32 accum.
// Q/K staged async; V staged transposed with packed dword stores.
// ---------------------------------------------------------------------------
__global__ __launch_bounds__(128) void attn_kernel(
    const unsigned short* __restrict__ qws,   // (B,S,1024) bf16
    const unsigned short* __restrict__ kvws,  // (B,G,2048) bf16  [k | v]
    const int* __restrict__ mask,             // (B,G) int32
    unsigned short* __restrict__ attnws)      // (B,S,1024) bf16
{
    constexpr int HD = 64, BM = 64, BG = 64, GT = 512, ST = 2048, E = 1024, PAD = 72;
    constexpr float SCALE = 0.125f;           // 64^-0.5

    __shared__ unsigned short Qs[BM * PAD];       // [s][d]
    __shared__ unsigned short Ks[BG * PAD];       // [g][d]
    __shared__ unsigned short Vt[HD * PAD];       // [d][g]  (transposed V)
    __shared__ unsigned short Ps[4 * 16 * PAD];   // per-wave 16x64 P scratch

    const int tid  = threadIdx.x;
    const int lane = tid & 31;
    const int wave = tid >> 5;       // 0..3 -> 16 query rows each
    const int half = (lane >> 4) & 1;
    const int l16  = lane & 15;

    const int sBase = blockIdx.x * BM;
    const int h     = blockIdx.y;
    const int b     = blockIdx.z;

    const unsigned short* qbase = qws  + (size_t)b * ST * E    + h * HD;
    const unsigned short* kbase = kvws + (size_t)b * GT * 2048 + h * HD;
    const unsigned short* vbase = kbase + 1024;

    // ---- stage Q tile once (512 chunks, 4 per thread) ----
    #pragma unroll
    for (int i = 0; i < 4; ++i) {
        int c = tid + i * 128;
        int r = c >> 3, cc = (c & 7) * 8;
        cp_lds_b128(&Qs[r * PAD + cc], &qbase[(size_t)(sBase + r) * E + cc]);
    }
    async_fence();
    __syncthreads();

    FragB qf0, qf1;
    #pragma unroll
    for (int vg = 0; vg < 8; ++vg) {
        qf0.u[vg] = *(const unsigned int*)&Qs[(wave * 16 + l16) * PAD + kpA(vg, half)];
        qf1.u[vg] = *(const unsigned int*)&Qs[(wave * 16 + l16) * PAD + 32 + kpA(vg, half)];
    }

    v8f   O[4] = {};
    float mrow[8], lrow[8];
    #pragma unroll
    for (int r = 0; r < 8; ++r) { mrow[r] = -3.40282e38f; lrow[r] = 0.0f; }

    for (int g0 = 0; g0 < GT; g0 += BG) {
        __syncthreads();             // prior tile fully consumed
        // K: async copy (512 chunks, 4 per thread)
        #pragma unroll
        for (int i = 0; i < 4; ++i) {
            int c = tid + i * 128;
            int g = c >> 3, cc = (c & 7) * 8;
            cp_lds_b128(&Ks[g * PAD + cc], &kbase[(size_t)(g0 + g) * 2048 + cc]);
        }
        // V: transposed staging with packed (g,g+1) dword stores (2 items/thread)
        #pragma unroll
        for (int i = 0; i < 2; ++i) {
            int c  = tid + i * 128;           // 0..255: 32 g-pairs x 8 d-chunks
            int gp = c >> 3, cc = (c & 7) * 8;
            int g  = gp * 2;
            U4 r0, r1;
            r0.q = *(const uint4*)&vbase[(size_t)(g0 + g)     * 2048 + cc];
            r1.q = *(const uint4*)&vbase[(size_t)(g0 + g + 1) * 2048 + cc];
            #pragma unroll
            for (int j = 0; j < 8; ++j) {
                unsigned int pk = (unsigned int)r0.s[j] | ((unsigned int)r1.s[j] << 16);
                *(unsigned int*)&Vt[(cc + j) * PAD + g] = pk;
            }
        }
        async_fence();
        __syncthreads();

        // ---- scores: S(16x64) = Q(16x64) * K^T ----
        v8f sf[4];
        #pragma unroll
        for (int f = 0; f < 4; ++f) {
            v8f acc = {};
            FragB kb0, kb1;
            #pragma unroll
            for (int vg = 0; vg < 8; ++vg) {
                kb0.u[vg] = *(const unsigned int*)&Ks[(f * 16 + l16) * PAD + kpB(vg, half)];
                kb1.u[vg] = *(const unsigned int*)&Ks[(f * 16 + l16) * PAD + 32 + kpB(vg, half)];
            }
            acc = wmma_bf16(qf0, kb0, acc);
            acc = wmma_bf16(qf1, kb1, acc);
            sf[f] = acc;
        }

        // ---- scale + mask ----
        int mk[4];
        #pragma unroll
        for (int f = 0; f < 4; ++f) mk[f] = mask[b * GT + g0 + f * 16 + l16];
        #pragma unroll
        for (int f = 0; f < 4; ++f)
            #pragma unroll
            for (int r = 0; r < 8; ++r)
                sf[f][r] = mk[f] ? sf[f][r] * SCALE : -3.40282e38f;

        // ---- online softmax, P to LDS, rescale O ----
        #pragma unroll
        for (int r = 0; r < 8; ++r) {
            float mx = fmaxf(fmaxf(sf[0][r], sf[1][r]), fmaxf(sf[2][r], sf[3][r]));
            #pragma unroll
            for (int off = 8; off; off >>= 1) mx = fmaxf(mx, __shfl_xor(mx, off, 32));
            float mnew  = fmaxf(mrow[r], mx);
            float alpha = __expf(mrow[r] - mnew);
            float psum  = 0.0f;
            #pragma unroll
            for (int f = 0; f < 4; ++f) {
                float p = __expf(sf[f][r] - mnew);
                sf[f][r] = p;
                psum += p;
            }
            #pragma unroll
            for (int off = 8; off; off >>= 1) psum += __shfl_xor(psum, off, 32);
            lrow[r] = lrow[r] * alpha + psum;
            mrow[r] = mnew;
            #pragma unroll
            for (int f2 = 0; f2 < 4; ++f2) O[f2][r] *= alpha;
            int rr = r + 8 * half;
            #pragma unroll
            for (int f = 0; f < 4; ++f)
                Ps[wave * (16 * PAD) + rr * PAD + f * 16 + l16] = f2bf(sf[f][r]);
        }

        // ---- P (C-layout) -> A-layout via per-wave LDS (same-wave DS in-order) ----
        FragB pa0, pa1;
        #pragma unroll
        for (int vg = 0; vg < 8; ++vg) {
            pa0.u[vg] = *(const unsigned int*)&Ps[wave * (16 * PAD) + l16 * PAD + kpA(vg, half)];
            pa1.u[vg] = *(const unsigned int*)&Ps[wave * (16 * PAD) + l16 * PAD + 32 + kpA(vg, half)];
        }

        // ---- O += P(16x64) * V(64x64); V B-frags are contiguous pair loads ----
        #pragma unroll
        for (int f2 = 0; f2 < 4; ++f2) {
            const int dcol = f2 * 16 + l16;
            FragB vb0, vb1;
            #pragma unroll
            for (int vg = 0; vg < 8; ++vg) {
                vb0.u[vg] = *(const unsigned int*)&Vt[dcol * PAD + kpB(vg, half)];
                vb1.u[vg] = *(const unsigned int*)&Vt[dcol * PAD + 32 + kpB(vg, half)];
            }
            O[f2] = wmma_bf16(pa0, vb0, O[f2]);
            O[f2] = wmma_bf16(pa1, vb1, O[f2]);
        }
    }

    // ---- normalize + write attn (B,S,E) bf16 ----
    #pragma unroll
    for (int f2 = 0; f2 < 4; ++f2) {
        int d = f2 * 16 + l16;
        #pragma unroll
        for (int r = 0; r < 8; ++r) {
            int s = sBase + wave * 16 + r + 8 * half;
            float inv = 1.0f / lrow[r];
            attnws[((size_t)b * ST + s) * E + h * HD + d] = f2bf(O[f2][r] * inv);
        }
    }
}

// ---------------------------------------------------------------------------
// Host-side launcher
// ---------------------------------------------------------------------------
extern "C" void kernel_launch(void* const* d_in, const int* in_sizes, int n_in,
                              void* d_out, int out_size, void* d_ws, size_t ws_size,
                              hipStream_t stream) {
    (void)in_sizes; (void)n_in; (void)out_size; (void)ws_size;

    const float* query   = (const float*)d_in[0];   // (8,2048,1024)
    const float* guide_v = (const float*)d_in[1];   // (8,512,768)
    const int*   amask   = (const int*)d_in[2];     // (8,512)
    const float* text_w  = (const float*)d_in[3];   // (768,1024)
    const float* text_b  = (const float*)d_in[4];
    const float* q_w     = (const float*)d_in[5];   // (1024,1024)
    const float* q_b     = (const float*)d_in[6];
    const float* kv_w    = (const float*)d_in[7];   // (1024,2048)
    const float* kv_b    = (const float*)d_in[8];
    const float* out_w   = (const float*)d_in[9];   // (1024,1024)
    const float* out_b   = (const float*)d_in[10];

    char* ws = (char*)d_ws;
    size_t off = 0;
    unsigned short* qc   = (unsigned short*)(ws + off); off += (size_t)16384 * 1024 * 2; // 32 MiB
    unsigned short* gc   = (unsigned short*)(ws + off); off += (size_t)4096  * 768  * 2; //  6 MiB
    unsigned short* twT  = (unsigned short*)(ws + off); off += (size_t)1024  * 768  * 2;
    unsigned short* qwT  = (unsigned short*)(ws + off); off += (size_t)1024  * 1024 * 2;
    unsigned short* kvwT = (unsigned short*)(ws + off); off += (size_t)2048  * 1024 * 2;
    unsigned short* owT  = (unsigned short*)(ws + off); off += (size_t)1024  * 1024 * 2;
    unsigned short* gws  = (unsigned short*)(ws + off); off += (size_t)4096  * 1024 * 2; // guide
    unsigned short* qws  = (unsigned short*)(ws + off); off += (size_t)16384 * 1024 * 2; // q proj
    unsigned short* kvws = (unsigned short*)(ws + off); off += (size_t)4096  * 2048 * 2; // kv proj
    unsigned short* aws  = (unsigned short*)(ws + off); off += (size_t)16384 * 1024 * 2; // attn out

    // 0) one-time conversions: activations -> bf16, weights -> bf16 transposed [N][K]
    convert_f32_bf16_kernel<<<16384, 256, 0, stream>>>(query,   qc, 16384 * 1024);
    convert_f32_bf16_kernel<<<3072,  256, 0, stream>>>(guide_v, gc, 4096 * 768);
    transpose_convert_kernel<<<dim3(1024 / 32,  768 / 32), dim3(32, 32), 0, stream>>>(text_w, twT,  768, 1024);
    transpose_convert_kernel<<<dim3(1024 / 32, 1024 / 32), dim3(32, 32), 0, stream>>>(q_w,   qwT, 1024, 1024);
    transpose_convert_kernel<<<dim3(2048 / 32, 1024 / 32), dim3(32, 32), 0, stream>>>(kv_w,  kvwT, 1024, 2048);
    transpose_convert_kernel<<<dim3(1024 / 32, 1024 / 32), dim3(32, 32), 0, stream>>>(out_w, owT, 1024, 1024);

    // 1) guide = guide_vector @ text_w + text_b            (4096 x 768 x 1024)
    gemm_bf16_kernel<unsigned short>
        <<<dim3(1024 / 128, 4096 / 128), 256, 0, stream>>>(gc, twT, text_b, gws, 4096, 768, 1024);

    // 2) q = query @ q_w + q_b                             (16384 x 1024 x 1024)
    gemm_bf16_kernel<unsigned short>
        <<<dim3(1024 / 128, 16384 / 128), 256, 0, stream>>>(qc, qwT, q_b, qws, 16384, 1024, 1024);

    // 3) kv = guide @ kv_w + kv_b                          (4096 x 1024 x 2048)
    gemm_bf16_kernel<unsigned short>
        <<<dim3(2048 / 128, 4096 / 128), 256, 0, stream>>>(gws, kvwT, kv_b, kvws, 4096, 1024, 2048);

    // 4) attention (flash, online softmax)
    attn_kernel<<<dim3(2048 / 64, 16, 8), 128, 0, stream>>>(qws, kvws, amask, aws);

    // 5) out = attn @ out_w + out_b -> fp32                (16384 x 1024 x 1024)
    gemm_bf16_kernel<float>
        <<<dim3(1024 / 128, 16384 / 128), 256, 0, stream>>>(aws, owT, out_b, (float*)d_out, 16384, 1024, 1024);
}
